// HeterGraphAttentionNetwork_6828998001378
// MI455X (gfx1250) — matrix operand order repaired and checked by reading, not compiled, not run on hardware
//
#include <hip/hip_runtime.h>
#include <hip/hip_bf16.h>

// Problem constants (B=1)
#define NN   4096
#define DD   64
#define HH   4
#define OO   64
#define KK   2
#define NUSER 4000
#define KH   (KK*HH)            // 8 (k,h) slices
#define LROW 136                // padded LDS row stride in bf16 elems (17*8 -> 16B aligned, bank-skewed)

typedef __attribute__((ext_vector_type(16))) __bf16        v16bf;
typedef __attribute__((ext_vector_type(8)))  float         v8f;
typedef __attribute__((ext_vector_type(4)))  unsigned int  v4u;

union FragU {
    v16bf bf;
    v4u   u[2];
    unsigned short us[16];
};

__device__ __forceinline__ unsigned short f2bf(float f) {
    unsigned int u = __float_as_uint(f);
    u += 0x7FFFu + ((u >> 16) & 1u);     // round-to-nearest-even
    return (unsigned short)(u >> 16);
}

// ---------------------------------------------------------------------------
// Kernel 1: h_prime = h @ w[k][h]  (WMMA bf16), then
//   hpT[kh][o][n] = bf16(h_prime)           (transposed, for attention B-frags)
//   src[kh][n]    = tanh(h_prime) . a_src
//   dst[kh][n]    = tanh(h_prime) . a_dst
// ---------------------------------------------------------------------------
__global__ void gat_hprime(const float* __restrict__ h,
                           const float* __restrict__ w,
                           const float* __restrict__ a_src,
                           const float* __restrict__ a_dst,
                           unsigned short* __restrict__ hpT,
                           float* __restrict__ src_g,
                           float* __restrict__ dst_g)
{
    const int kh  = blockIdx.x >> 8;            // 0..7
    const int n0  = (blockIdx.x & 255) << 4;    // row tile base
    const int tid = threadIdx.x;
    const int wv  = tid >> 5;
    const int lane = tid & 31;

    __shared__ float sS[16];
    __shared__ float sD[16];
    if (tid < 16) { sS[tid] = 0.f; sD[tid] = 0.f; }
    __syncthreads();

    const float* wk = w + (size_t)kh * DD * OO;     // w[k][h][f][o]
    const int arow  = n0 + (lane & 15);             // A-matrix row (node index)
    const int ocol  = (wv << 4) + (lane & 15);      // output column o

    v8f acc = {};
    #pragma unroll
    for (int c = 0; c < 2; ++c) {                   // K chunks of 32 (D=64)
        FragU a, b;
        // A fragment: lanes 0-15 hold K 0..15, lanes 16-31 hold K 16..31
        const int ka = 32 * c + ((lane < 16) ? 0 : 16);
        #pragma unroll
        for (int half = 0; half < 2; ++half) {
            const int base = ka + 8 * half;
            #pragma unroll
            for (int j = 0; j < 8; ++j)
                a.us[half * 8 + j] = f2bf(h[(size_t)arow * DD + base + j]);
        }
        // B fragment: lanes 0-15 hold K 0..15, lanes 16-31 hold K 16..31 (column ocol)
        const int kb = 32 * c + ((lane < 16) ? 0 : 16);
        #pragma unroll
        for (int j = 0; j < 16; ++j)
            b.us[j] = f2bf(wk[(size_t)(kb + j) * OO + ocol]);

        acc = __builtin_amdgcn_wmma_f32_16x16x32_bf16(false, a.bf, false, b.bf,
                                                      (short)0, acc, false, false);
    }

    const float aS = a_src[kh * OO + ocol];
    const float aD = a_dst[kh * OO + ocol];
    #pragma unroll
    for (int r = 0; r < 8; ++r) {
        const int m  = r + ((lane < 16) ? 0 : 8);   // C/D layout: VGPR r -> M=r (+8 for hi lanes)
        const float hp = acc[r];
        hpT[((size_t)kh * OO + ocol) * NN + n0 + m] = f2bf(hp);
        const float t = tanhf(hp);
        atomicAdd(&sS[m], t * aS);
        atomicAdd(&sD[m], t * aD);
    }
    __syncthreads();
    if (tid < 16) {
        src_g[kh * NN + n0 + tid] = sS[tid];
        dst_g[kh * NN + n0 + tid] = sD[tid];
    }
}

// ---------------------------------------------------------------------------
// Kernel 2: fused masked-softmax attention * h_prime (flash-style; scores are
// tanh-bounded so exp without max subtraction is numerically safe).
// block = (k,h, 16-row tile); 128 threads; 32 column tiles of 128.
// V tiles stream via global_load_async_to_lds_b128 (ASYNCcnt), double-buffered.
// ---------------------------------------------------------------------------
__global__ void __launch_bounds__(128)
gat_attn(const float* __restrict__ hadj,
         const unsigned short* __restrict__ hpT,
         const float* __restrict__ src_g,
         const float* __restrict__ dst_g,
         float* __restrict__ gat)
{
    // tile-major block index: the 4 heads of the same adjacency slab run together
    const int kh  = blockIdx.x & 7;
    const int n0  = (blockIdx.x >> 3) << 4;
    const int k   = kh >> 2;
    const float* adj = hadj + (size_t)k * NN * NN;
    const int tid  = threadIdx.x;
    const int wv   = tid >> 5;
    const int lane = tid & 31;

    __shared__ __align__(16) unsigned short sP[16 * LROW];       // exp-scores (bf16)
    __shared__ __align__(16) unsigned short sV[2][64 * LROW];    // h_prime tiles [o][j] (bf16), double-buffered
    __shared__ float sSrc[16];
    __shared__ float sRow[16];

    if (tid < 16) {
        sSrc[tid] = src_g[kh * NN + n0 + tid];
        sRow[tid] = 0.f;
    }

    float rsum[16];
    #pragma unroll
    for (int r = 0; r < 16; ++r) rsum[r] = 0.f;
    v8f acc = {};

    // global base of this (k,h) slice of hpT (bytes); per-thread 32-bit offsets
    const unsigned long long vbase =
        (unsigned long long)hpT + (unsigned long long)kh * OO * NN * 2ull;

    const int arow  = lane & 15;
    const int ahalf = (lane < 16) ? 0 : 8;
    const int brow  = (wv << 4) + (lane & 15);      // o column owned by this wave/lane
    const int bhalf = (lane < 16) ? 0 : 8;

    // ---- async-stage V tile jt into buffer (jt&1): 64 rows x 256B, 16B chunks ----
    auto stage_tile = [&](int jt) {
        const int buf = jt & 1;
        const unsigned jbyte = (unsigned)(jt << 8);          // j0*2 bytes
        #pragma unroll
        for (int i = 0; i < 8; ++i) {
            const int idx = tid + (i << 7);                  // 0..1023
            const int o   = idx >> 4;
            const int c   = idx & 15;
            const unsigned voff = (unsigned)o * (NN * 2u) + jbyte + ((unsigned)c << 4);
            const unsigned ldsa =
                (unsigned)(unsigned long long)&sV[buf][o * LROW + (c << 3)];
            asm volatile("global_load_async_to_lds_b128 %0, %1, %2"
                         :: "v"(ldsa), "v"(voff), "s"(vbase) : "memory");
        }
    };

    stage_tile(0);   // prologue

    for (int jt = 0; jt < 32; ++jt) {
        const int j0  = jt << 7;
        const int cur = jt & 1;
        __syncthreads();   // previous iteration's WMMA reads of sP / sV[cur] complete

        // ---- masked exp-scores for column j0+tid, rows 0..15 (f32 VALU) ----
        const float d = dst_g[kh * NN + j0 + tid];
        #pragma unroll
        for (int r = 0; r < 16; ++r) {
            const float* ap = adj + (size_t)(n0 + r) * NN + j0 + tid;
            __builtin_prefetch(ap + 128, 0, 0);     // next column tile (speculative)
            const float a = *ap;
            float s = sSrc[r] + d;
            s = (s >= 0.f) ? s : 0.2f * s;          // LeakyReLU(0.2)
            const float e = (a > 0.f) ? __expf(s) : 0.f;
            rsum[r] += e;
            sP[r * LROW + tid] = f2bf(e);
        }

        // ---- kick off next V tile, then drain current tile's async loads ----
        if (jt + 1 < 32) {
            stage_tile(jt + 1);
            asm volatile("s_wait_asynccnt 0x8" ::: "memory");   // tile jt resident, jt+1 in flight
        } else {
            asm volatile("s_wait_asynccnt 0x0" ::: "memory");
        }
        __syncthreads();   // sP ready + everyone's async data visible

        // ---- accumulate P(16x128) x V(128x16) via 4 bf16 WMMAs ----
        const unsigned short* sVc = sV[cur];
        #pragma unroll
        for (int kc = 0; kc < 4; ++kc) {
            const int kb = kc << 5;
            FragU a, b;
            a.u[0] = *(const v4u*)&sP[arow * LROW + kb + ahalf];
            a.u[1] = *(const v4u*)&sP[arow * LROW + kb + 16 + ahalf];
            b.u[0] = *(const v4u*)&sVc[brow * LROW + kb + bhalf];
            b.u[1] = *(const v4u*)&sVc[brow * LROW + kb + 16 + bhalf];
            acc = __builtin_amdgcn_wmma_f32_16x16x32_bf16(false, a.bf, false, b.bf,
                                                          (short)0, acc, false, false);
        }
    }

    __syncthreads();
    #pragma unroll
    for (int r = 0; r < 16; ++r) atomicAdd(&sRow[r], rsum[r]);
    __syncthreads();

    #pragma unroll
    for (int r = 0; r < 8; ++r) {
        const int m = r + ((lane < 16) ? 0 : 8);
        const float v = acc[r] / sRow[m];           // softmax normalization
        gat[(((size_t)kh * NN) + n0 + m) * OO + brow] = v;
    }
}

// ---------------------------------------------------------------------------
// Kernel 3: mean over heads, concat over k, FC (C=2), log_softmax.
// One wave per user node.
// ---------------------------------------------------------------------------
__global__ void gat_head(const float* __restrict__ gat,
                         const float* __restrict__ fc_w,
                         const float* __restrict__ fc_b,
                         float* __restrict__ out)
{
    const int wid  = blockIdx.x * (blockDim.x >> 5) + (threadIdx.x >> 5);
    const int lane = threadIdx.x & 31;
    if (wid >= NUSER) return;

    float l0 = 0.f, l1 = 0.f;
    #pragma unroll
    for (int f = lane; f < KK * OO; f += 32) {
        const int k = f >> 6;
        const int o = f & 63;
        float s = 0.f;
        #pragma unroll
        for (int hh = 0; hh < HH; ++hh)
            s += gat[(((size_t)(k * HH + hh) * NN) + wid) * OO + o];
        s *= 0.25f;                                  // mean over heads
        l0 += s * fc_w[f];
        l1 += s * fc_w[KK * OO + f];
    }
    #pragma unroll
    for (int off = 16; off > 0; off >>= 1) {
        l0 += __shfl_down(l0, off);
        l1 += __shfl_down(l1, off);
    }
    if (lane == 0) {
        l0 += fc_b[0];
        l1 += fc_b[1];
        const float m = fmaxf(l0, l1);
        const float z = logf(__expf(l0 - m) + __expf(l1 - m));
        out[(size_t)wid * 2 + 0] = l0 - m - z;
        out[(size_t)wid * 2 + 1] = l1 - m - z;
    }
}

// ---------------------------------------------------------------------------
extern "C" void kernel_launch(void* const* d_in, const int* in_sizes, int n_in,
                              void* d_out, int out_size, void* d_ws, size_t ws_size,
                              hipStream_t stream)
{
    const float* h     = (const float*)d_in[0];
    const float* hadj  = (const float*)d_in[1];
    const float* w     = (const float*)d_in[2];
    const float* a_src = (const float*)d_in[3];
    const float* a_dst = (const float*)d_in[4];
    const float* fc_w  = (const float*)d_in[5];
    const float* fc_b  = (const float*)d_in[6];
    float* out = (float*)d_out;

    char* ws = (char*)d_ws;
    unsigned short* hpT = (unsigned short*)ws;                         // KH*OO*NN bf16 = 4 MB
    float* src_g = (float*)(ws + (size_t)(4u << 20));                  // KH*NN f32 = 128 KB
    float* dst_g = (float*)(ws + (size_t)(4u << 20) + (128u << 10));   // 128 KB
    float* gat   = (float*)(ws + (size_t)(4u << 20) + (256u << 10));   // KH*NN*OO f32 = 8 MB

    (void)in_sizes; (void)n_in; (void)out_size; (void)ws_size;

    dim3 blk(128);
    dim3 grd(KH * (NN / 16));   // 2048
    gat_hprime<<<grd, blk, 0, stream>>>(h, w, a_src, a_dst, hpT, src_g, dst_g);
    gat_attn<<<grd, blk, 0, stream>>>(hadj, hpT, src_g, dst_g, gat);
    gat_head<<<500, 256, 0, stream>>>(gat, fc_w, fc_b, out);
}
